// DKVMN_51625506898541
// MI455X (gfx1250) — matrix operand order repaired and checked by reading, not compiled, not run on hardware
//
#include <hip/hip_runtime.h>
#include <hip/hip_bf16.h>
#include <math.h>

typedef _Float16 h16;
typedef h16  v16h __attribute__((ext_vector_type(16)));
typedef h16  v8h  __attribute__((ext_vector_type(8)));
typedef float v8f __attribute__((ext_vector_type(8)));

#define SB    102400   // S*B = 200*512
#define BATCH 512
#define SEQ   200
#define MSZ   64
#define QD    128
#define QAD   256
#define KPAD  320      // 256 reads + 50 inp_fc + 14 zero pad
#define FC    512

// ---------------- WMMA helpers (16x16x32 f16 -> f32) ----------------
__device__ inline v8f zero8() { v8f z = {0.f,0.f,0.f,0.f,0.f,0.f,0.f,0.f}; return z; }

// A fragment: 16x32 tile at (row0, k0), row-major A with leading dim lda (halves).
// lane<16: rows 0..15, elems0-7 = K k0..k0+7, elems8-15 = K k0+16..k0+23
// lane>=16: elems0-7 = K k0+8..15, elems8-15 = K k0+24..31
__device__ inline v16h load_a_frag(const h16* A, int lda, int row0, int k0, int lane) {
    int r = lane & 15, hi = lane >> 4;
    const h16* p = A + (size_t)(row0 + r) * lda + k0 + hi * 8;
    v16h a;
    *(v8h*)&a         = *(const v8h*)p;
    *(((v8h*)&a) + 1) = *(const v8h*)(p + 16);
    return a;
}

// B fragment: 32x16 tile at (k0, n0) from B stored column-major-as-rows:
// Bt[n*ldb + k] = B[k][n]. lane column n0+(lane&15); lanes<16: elems0-7 = K k0..k0+7,
// elems8-15 = K k0+16..k0+23; lanes>=16: K k0+8..15, k0+24..31 (same as A).
__device__ inline v16h load_b_frag(const h16* Bt, int ldb, int n0, int k0, int lane) {
    int n = n0 + (lane & 15), hi = lane >> 4;
    const h16* p = Bt + (size_t)n * ldb + k0 + hi * 8;
    v16h b;
    *(v8h*)&b         = *(const v8h*)p;
    *(((v8h*)&b) + 1) = *(const v8h*)(p + 16);
    return b;
}

__device__ inline v8f wmma16(v16h a, v16h b, v8f c) {
    return __builtin_amdgcn_wmma_f32_16x16x32_f16(false, a, false, b, (short)0, c, false, false);
}

// ---------------- weight prep ----------------
__global__ void k_cast(const float* __restrict__ src, h16* __restrict__ dst, int n) {
    int i = blockIdx.x * 256 + threadIdx.x;
    if (i < n) dst[i] = (h16)src[i];
}

// dst[n*Kp + k] = (k < K) ? src[k*N + n] : 0   (transpose + K pad, f32->f16)
__global__ void k_transpose_pad(const float* __restrict__ src, h16* __restrict__ dst,
                                int K, int N, int Kp) {
    int i = blockIdx.x * 256 + threadIdx.x;
    if (i >= N * Kp) return;
    int n = i / Kp, k = i - n * Kp;
    dst[i] = (h16)(k < K ? src[(size_t)k * N + n] : 0.f);
}

// ---------------- embeddings ----------------
__global__ void k_embed(const int* __restrict__ qd, const int* __restrict__ qad,
                        const int* __restrict__ pidd,
                        const float* __restrict__ qe, const float* __restrict__ qae,
                        const float* __restrict__ qed, const float* __restrict__ qaed,
                        const float* __restrict__ diffp,
                        h16* __restrict__ qeh, h16* __restrict__ qah) {
    int idx = blockIdx.x;        // 0..SB-1 (= s*B + b)
    int d = threadIdx.x;         // 0..127
    int q = qd[idx], qa = qad[idx], pid = pidd[idx];
    float p = diffp[pid];
    float v = qe[(size_t)q * QD + d] + p * qed[(size_t)q * QD + d];
    qeh[(size_t)idx * QD + d] = (h16)v;
    for (int dd = d; dd < QAD; dd += QD) {
        float u = qae[(size_t)qa * QAD + dd] + p * qaed[(size_t)qa * QAD + dd];
        qah[(size_t)idx * QAD + dd] = (h16)u;
    }
}

// pid^2 regularization sum (deterministic tree reduce)
__global__ void k_regloss(const int* __restrict__ pidd, const float* __restrict__ diffp,
                          float* __restrict__ reg) {
    __shared__ float sh[256];
    int t = threadIdx.x;
    float s = 0.f;
    for (int i = t; i < SB; i += 256) { float p = diffp[pidd[i]]; s += p * p; }
    sh[t] = s; __syncthreads();
    for (int o = 128; o > 0; o >>= 1) { if (t < o) sh[t] += sh[t + o]; __syncthreads(); }
    if (t == 0) reg[0] = sh[0] * 1e-5f;
}

// ---------------- attention logits + fused softmax ----------------
// logits(SB x 64) = qeh(SB x 128) @ mem_key^T ; softmax over 64 -> w (f32)
__global__ void __launch_bounds__(32) k_attn(const h16* __restrict__ qeh,
                                             const h16* __restrict__ mkh,
                                             float* __restrict__ wout) {
    int lane = threadIdx.x;
    int row0 = blockIdx.x * 16;
    v8f acc0 = zero8(), acc1 = zero8(), acc2 = zero8(), acc3 = zero8();
#pragma unroll
    for (int ks = 0; ks < 4; ++ks) {
        v16h a = load_a_frag(qeh, QD, row0, ks * 32, lane);
        acc0 = wmma16(a, load_b_frag(mkh, QD,  0, ks * 32, lane), acc0);
        acc1 = wmma16(a, load_b_frag(mkh, QD, 16, ks * 32, lane), acc1);
        acc2 = wmma16(a, load_b_frag(mkh, QD, 32, ks * 32, lane), acc2);
        acc3 = wmma16(a, load_b_frag(mkh, QD, 48, ks * 32, lane), acc3);
    }
    int nl = lane & 15, mb = (lane >> 4) * 8;
#pragma unroll
    for (int i = 0; i < 8; ++i) {
        float v0 = acc0[i], v1 = acc1[i], v2 = acc2[i], v3 = acc3[i];
        float m = fmaxf(fmaxf(v0, v1), fmaxf(v2, v3));
        for (int msk = 1; msk < 16; msk <<= 1) m = fmaxf(m, __shfl_xor(m, msk, 32));
        float e0 = expf(v0 - m), e1 = expf(v1 - m), e2 = expf(v2 - m), e3 = expf(v3 - m);
        float s = e0 + e1 + e2 + e3;
        for (int msk = 1; msk < 16; msk <<= 1) s += __shfl_xor(s, msk, 32);
        float inv = 1.f / s;
        size_t base = (size_t)(row0 + mb + i) * MSZ;
        wout[base +  0 + nl] = e0 * inv;
        wout[base + 16 + nl] = e1 * inv;
        wout[base + 32 + nl] = e2 * inv;
        wout[base + 48 + nl] = e3 * inv;
    }
}

// ---------------- erase + add GEMMs, fused (shared A fragments) ----------------
// One wave computes a 16-row x 32-col tile for BOTH erase (We) and add (Wa):
// 1 A-fragment load feeds 4 WMMAs per k-step.
__global__ void __launch_bounds__(32) k_ea(const h16* __restrict__ qah,
                                           const h16* __restrict__ weh,
                                           const h16* __restrict__ wah,
                                           const float* __restrict__ be,
                                           const float* __restrict__ ba,
                                           float* __restrict__ E, float* __restrict__ Aad) {
    int lane = threadIdx.x;
    int row0 = blockIdx.x * 16;
    int n0 = blockIdx.y * 32;
    v8f ce0 = zero8(), ce1 = zero8(), ca0 = zero8(), ca1 = zero8();
#pragma unroll
    for (int ks = 0; ks < 8; ++ks) {
        v16h a = load_a_frag(qah, QAD, row0, ks * 32, lane);
        ce0 = wmma16(a, load_b_frag(weh, QAD, n0,      ks * 32, lane), ce0);
        ce1 = wmma16(a, load_b_frag(weh, QAD, n0 + 16, ks * 32, lane), ce1);
        ca0 = wmma16(a, load_b_frag(wah, QAD, n0,      ks * 32, lane), ca0);
        ca1 = wmma16(a, load_b_frag(wah, QAD, n0 + 16, ks * 32, lane), ca1);
    }
    int na = n0 + (lane & 15), nb = na + 16, mb = (lane >> 4) * 8;
    float bea = be[na], beb = be[nb], baa = ba[na], bab = ba[nb];
#pragma unroll
    for (int i = 0; i < 8; ++i) {
        size_t rbase = (size_t)(row0 + mb + i) * QAD;
        E[rbase + na]   = 1.f / (1.f + expf(-(ce0[i] + bea)));
        E[rbase + nb]   = 1.f / (1.f + expf(-(ce1[i] + beb)));
        Aad[rbase + na] = tanhf(ca0[i] + baa);
        Aad[rbase + nb] = tanhf(ca1[i] + bab);
    }
}

// ---------------- sequential memory scan ----------------
// one block per batch element; thread d owns column d of the 64x256 memory in regs
__global__ void __launch_bounds__(256) k_scan(const float* __restrict__ mem_init,
                                              const float* __restrict__ w,
                                              const float* __restrict__ E,
                                              const float* __restrict__ Aad,
                                              h16* __restrict__ feath) {
    int b = blockIdx.x;
    int d = threadIdx.x;
    float memv[MSZ];
#pragma unroll
    for (int m = 0; m < MSZ; ++m) memv[m] = mem_init[m * QAD + d];
    __shared__ float wl[MSZ];
    for (int s = 0; s < SEQ; ++s) {
        int idx = s * BATCH + b;
        if (d < MSZ) wl[d] = w[(size_t)idx * MSZ + d];
        __syncthreads();
        float e = E[(size_t)idx * QAD + d];
        float a = Aad[(size_t)idx * QAD + d];
        float acc = 0.f;
#pragma unroll
        for (int m = 0; m < MSZ; ++m) {
            float wm = wl[m];
            acc += wm * memv[m];
            memv[m] = memv[m] * (1.f - wm * e) + wm * a;
        }
        feath[(size_t)idx * KPAD + d] = (h16)acc;   // feat cols 0..255 = reads
        __syncthreads();
    }
}

// ---------------- inp_fc = tanh(q_e @ Wi + bi), feat cols 256..305 (pad to 319) ----
__global__ void __launch_bounds__(256) k_inpfc(const h16* __restrict__ qeh,
                                               const float* __restrict__ Wi,
                                               const float* __restrict__ bi,
                                               h16* __restrict__ feath) {
    __shared__ float q[4][QD];
    int t = threadIdx.x;
    int rbase = blockIdx.x * 4;
    for (int i = t; i < 4 * QD; i += 256) {
        int r = i >> 7, k = i & 127;
        q[r][k] = (float)qeh[(size_t)(rbase + r) * QD + k];
    }
    __syncthreads();
    int r = t >> 6, j = t & 63;
    int row = rbase + r;
    h16 out = (h16)0.f;
    if (j < 50) {
        float s = bi[j];
        for (int k = 0; k < QD; ++k) s += q[r][k] * Wi[k * 50 + j];
        out = (h16)tanhf(s);
    }
    feath[(size_t)row * KPAD + QAD + j] = out;
}

// ---------------- final GEMM fused with tanh, Wo2 dot, sigmoid ----------------
// Each wave preloads its entire 16x320 A panel into registers (10 frags = 80 VGPRs)
// then streams only B fragments across its 8 column tiles.
__global__ void __launch_bounds__(128) k_final(const h16* __restrict__ feath,
                                               const h16* __restrict__ wo1h,
                                               const float* __restrict__ bo1,
                                               const float* __restrict__ Wo2,
                                               const float* __restrict__ bo2,
                                               float* __restrict__ predl,
                                               float* __restrict__ outsig) {
    int t = threadIdx.x, lane = t & 31, wave = t >> 5;
    int row0 = blockIdx.x * 16;

    v16h afrag[10];
#pragma unroll
    for (int ks = 0; ks < 10; ++ks)
        afrag[ks] = load_a_frag(feath, KPAD, row0, ks * 32, lane);

    float pp[8] = {0.f, 0.f, 0.f, 0.f, 0.f, 0.f, 0.f, 0.f};
    for (int c8 = 0; c8 < 8; ++c8) {
        int n0 = (wave * 8 + c8) * 16;
        v8f c = zero8();
#pragma unroll
        for (int ks = 0; ks < 10; ++ks) {
            v16h b = load_b_frag(wo1h, KPAD, n0, ks * 32, lane);
            c = wmma16(afrag[ks], b, c);
        }
        int n = n0 + (lane & 15);
        float bias = bo1[n], w2 = Wo2[n];
#pragma unroll
        for (int i = 0; i < 8; ++i) pp[i] += tanhf(c[i] + bias) * w2;
    }
#pragma unroll
    for (int i = 0; i < 8; ++i)
        for (int msk = 1; msk < 16; msk <<= 1) pp[i] += __shfl_xor(pp[i], msk, 32);
    __shared__ float red[4][2][8];
    if ((lane & 15) == 0) {
        int h = lane >> 4;
        for (int i = 0; i < 8; ++i) red[wave][h][i] = pp[i];
    }
    __syncthreads();
    if (t < 16) {
        float s = bo2[0];
        for (int wv = 0; wv < 4; ++wv) s += red[wv][t >> 3][t & 7];
        int row = row0 + t;
        predl[row] = s;
        outsig[row] = 1.f / (1.f + expf(-s));
    }
}

// ---------------- masked BCE loss (deterministic tree reduce) ----------------
__global__ void k_loss(const float* __restrict__ predl, const int* __restrict__ target,
                       const float* __restrict__ reg, float* __restrict__ dout) {
    __shared__ float sh[256];
    __shared__ int shc[256];
    int t = threadIdx.x;
    float s = 0.f; int c = 0;
    for (int i = t; i < SB; i += 256) {
        int lab = target[i];
        if (lab != -1) {
            float p = predl[i], y = (float)lab;
            s += fmaxf(p, 0.f) - p * y + log1pf(expf(-fabsf(p)));
            c++;
        }
    }
    sh[t] = s; shc[t] = c; __syncthreads();
    for (int o = 128; o > 0; o >>= 1) {
        if (t < o) { sh[t] += sh[t + o]; shc[t] += shc[t + o]; }
        __syncthreads();
    }
    if (t == 0) { dout[0] = sh[0] + reg[0]; dout[1 + SB] = (float)shc[0]; }
}

// ---------------- host launch ----------------
extern "C" void kernel_launch(void* const* d_in, const int* in_sizes, int n_in,
                              void* d_out, int out_size, void* d_ws, size_t ws_size,
                              hipStream_t stream) {
    const int*   q_data   = (const int*)d_in[0];
    const int*   qa_data  = (const int*)d_in[1];
    const int*   target   = (const int*)d_in[2];
    const int*   pid_data = (const int*)d_in[3];
    const float* mem_key  = (const float*)d_in[4];
    const float* mem_val0 = (const float*)d_in[5];
    const float* q_embed  = (const float*)d_in[6];
    const float* qa_embed = (const float*)d_in[7];
    const float* q_ediff  = (const float*)d_in[8];
    const float* qa_ediff = (const float*)d_in[9];
    const float* diff_p   = (const float*)d_in[10];
    const float* We  = (const float*)d_in[11];
    const float* be  = (const float*)d_in[12];
    const float* Wa  = (const float*)d_in[13];
    const float* ba  = (const float*)d_in[14];
    const float* Wi  = (const float*)d_in[15];
    const float* bi  = (const float*)d_in[16];
    const float* Wo1 = (const float*)d_in[17];
    const float* bo1 = (const float*)d_in[18];
    const float* Wo2 = (const float*)d_in[19];
    const float* bo2 = (const float*)d_in[20];
    float* out = (float*)d_out;

    char* ws = (char*)d_ws;
    size_t off = 0;
    auto alloc = [&](size_t bytes) -> void* {
        void* p = ws + off;
        off += (bytes + 255) & ~(size_t)255;
        return p;
    };
    h16*  mkh   = (h16*)alloc((size_t)MSZ * QD * 2);          // mem_key f16 (B layout [n][k])
    h16*  weh   = (h16*)alloc((size_t)QAD * QAD * 2);         // We^T f16
    h16*  wah   = (h16*)alloc((size_t)QAD * QAD * 2);         // Wa^T f16
    h16*  wo1h  = (h16*)alloc((size_t)FC * KPAD * 2);         // Wo1^T f16, K padded to 320
    h16*  qeh   = (h16*)alloc((size_t)SB * QD * 2);
    h16*  qah   = (h16*)alloc((size_t)SB * QAD * 2);
    float* wsm  = (float*)alloc((size_t)SB * MSZ * 4);        // softmax weights
    float* E    = (float*)alloc((size_t)SB * QAD * 4);        // erase
    float* Aad  = (float*)alloc((size_t)SB * QAD * 4);        // add
    h16*  feath = (h16*)alloc((size_t)SB * KPAD * 2);         // [reads | inp_fc | pad]
    float* predl = (float*)alloc((size_t)SB * 4);             // logits
    float* reg  = (float*)alloc(256);                         // reg-loss scalar

    // weight prep
    k_cast<<<(MSZ * QD + 255) / 256, 256, 0, stream>>>(mem_key, mkh, MSZ * QD);
    k_transpose_pad<<<(QAD * QAD + 255) / 256, 256, 0, stream>>>(We, weh, QAD, QAD, QAD);
    k_transpose_pad<<<(QAD * QAD + 255) / 256, 256, 0, stream>>>(Wa, wah, QAD, QAD, QAD);
    k_transpose_pad<<<(FC * KPAD + 255) / 256, 256, 0, stream>>>(Wo1, wo1h, QAD + 50, FC, KPAD);

    // embeddings + reg loss
    k_embed<<<SB, 128, 0, stream>>>(q_data, qa_data, pid_data, q_embed, qa_embed,
                                    q_ediff, qa_ediff, diff_p, qeh, qah);
    k_regloss<<<1, 256, 0, stream>>>(pid_data, diff_p, reg);

    // batched GEMMs (hoisted out of the scan)
    k_attn<<<SB / 16, 32, 0, stream>>>(qeh, mkh, wsm);
    k_ea<<<dim3(SB / 16, QAD / 32), 32, 0, stream>>>(qah, weh, wah, be, ba, E, Aad);

    // sequential state update (elementwise only)
    k_scan<<<BATCH, 256, 0, stream>>>(mem_val0, wsm, E, Aad, feath);

    // inp_fc into feat cols 256..319
    k_inpfc<<<SB / 4, 256, 0, stream>>>(qeh, Wi, bi, feath);

    // final GEMM fused with tanh + Wo2 + sigmoid
    k_final<<<SB / 16, 128, 0, stream>>>(feath, wo1h, bo1, Wo2, bo2, predl, out + 1);

    // loss + mask count
    k_loss<<<1, 256, 0, stream>>>(predl, target, reg, out);
}
